// MultiHeadLatentAttention_17875653886560
// MI455X (gfx1250) — matrix operand order repaired and checked by reading, compiled, not run on hardware
//
#include <hip/hip_runtime.h>
#include <hip/hip_bf16.h>

// ---------------------------------------------------------------------------
// MLA forward for MI455X (gfx1250, wave32, WMMA bf16 w/ fp32 accumulate)
// B=2 S=2048 E=2048 H=16 KV_RANK=512 ROPE=64 NOPE=64 VD=128 QD=128
// GEMM tiles staged with CDNA5 async copies (GLOBAL_LOAD_ASYNC_TO_LDS_B128,
// ASYNCcnt-tracked, double buffered).
// ---------------------------------------------------------------------------

typedef __attribute__((ext_vector_type(16))) __bf16 v16bf;
typedef __attribute__((ext_vector_type(8)))  float  v8f;

#define CB 2
#define CS 2048
#define CE 2048
#define CH 16
#define CKV 512
#define CROPE 64
#define CVD 128
#define CQD 128

__device__ __forceinline__ unsigned short f32_to_bf16(float f) {
    union { float f; unsigned u; } c; c.f = f;
    unsigned u = c.u;
    unsigned r = u + 0x7FFFu + ((u >> 16) & 1u);   // round-to-nearest-even
    return (unsigned short)(r >> 16);
}

// ---- CDNA5 async copy: global -> LDS, 16B per lane, ASYNCcnt tracked ------
__device__ __forceinline__ void async_b128(unsigned lds_off, const void* gsrc) {
    asm volatile("global_load_async_to_lds_b128 %0, %1, off"
                 :: "v"(lds_off), "v"(gsrc) : "memory");
}
__device__ __forceinline__ void async_b128_o16(unsigned lds_off, const void* gsrc) {
    // INST_OFFSET is added to BOTH the LDS and global addresses (ISA 10. async)
    asm volatile("global_load_async_to_lds_b128 %0, %1, off offset:16"
                 :: "v"(lds_off), "v"(gsrc) : "memory");
}
__device__ __forceinline__ void wait_async0() {
    asm volatile("s_wait_asynccnt 0x0" ::: "memory");
}

// ---------------------------------------------------------------------------
// fp32 -> bf16 conversion, 4 elements / thread
// ---------------------------------------------------------------------------
__global__ __launch_bounds__(256)
void cvt_f32_bf16_x4(const float* __restrict__ s, unsigned short* __restrict__ d, long n4) {
    long i = (long)blockIdx.x * 256 + threadIdx.x;
    if (i < n4) {
        float4 v = ((const float4*)s)[i];
        uint2 o;
        o.x = (unsigned)f32_to_bf16(v.x) | ((unsigned)f32_to_bf16(v.y) << 16);
        o.y = (unsigned)f32_to_bf16(v.z) | ((unsigned)f32_to_bf16(v.w) << 16);
        ((uint2*)d)[i] = o;
    }
}

// ---------------------------------------------------------------------------
// Generic WMMA GEMM: C[M,N] = A[M,K] (row major, lda) * B[N,K]^T (row major, ldb)
// bf16 inputs, fp32 accumulate, output bf16 or fp32 (ldc).
// Block: 256 threads = 8 waves, tile 128x128, BK=32. Waves: 4(M) x 2(N),
// each wave computes 2x4 = 8 WMMA 16x16 tiles (32x64 region).
// Tiles staged with async DMA, double buffered. N edge-guarded (N=576 case).
// ---------------------------------------------------------------------------
template<bool OUT_F32>
__global__ __launch_bounds__(256)
void gemm_bf16_nt(const unsigned short* __restrict__ A,
                  const unsigned short* __restrict__ Bm,
                  void* __restrict__ C,
                  int M, int N, int K, int lda, int ldb, int ldc)
{
    constexpr int BK = 32;
    constexpr int LSTR = 40;                 // LDS row stride (ushorts), 80B, conflict pad
    __shared__ __align__(16) unsigned short sA[2][128 * LSTR];
    __shared__ __align__(16) unsigned short sB[2][128 * LSTR];

    const int tid  = threadIdx.x;
    const int lane = tid & 31;
    const int wv   = tid >> 5;
    const int wm   = wv >> 1;                // 0..3
    const int wn   = wv & 1;                 // 0..1
    const int m0   = blockIdx.y * 128;
    const int n0   = blockIdx.x * 128;
    const int half = lane >> 4;
    const int l16  = lane & 15;

    const int ldRow = tid >> 1;              // 0..127
    const int ldK   = (tid & 1) * 16;        // 0 or 16

    v8f acc[2][4];
    for (int i = 0; i < 2; i++)
        for (int j = 0; j < 4; j++)
            for (int e = 0; e < 8; e++) acc[i][j][e] = 0.0f;

    union F { uint4 u[2]; v16bf v; };

    // issue async DMA for one 128x32 A tile + B tile into buffer `buf`
    auto stage = [&](int buf, int k0) {
        {
            const unsigned short* src = A + (size_t)(m0 + ldRow) * lda + k0 + ldK;
            unsigned dst = (unsigned)(size_t)&sA[buf][ldRow * LSTR + ldK];
            async_b128(dst, src);
            async_b128_o16(dst, src);
            if (k0 + BK < K) __builtin_prefetch(src + BK, 0, 1);   // global_prefetch_b8
        }
        int n = n0 + ldRow;
        if (n < N) {
            const unsigned short* src = Bm + (size_t)n * ldb + k0 + ldK;
            unsigned dst = (unsigned)(size_t)&sB[buf][ldRow * LSTR + ldK];
            async_b128(dst, src);
            async_b128_o16(dst, src);
        } else {
            uint4 z = make_uint4(0, 0, 0, 0);
            *(uint4*)(&sB[buf][ldRow * LSTR + ldK])     = z;
            *(uint4*)(&sB[buf][ldRow * LSTR + ldK + 8]) = z;
        }
    };

    const int nsteps = K / BK;
    stage(0, 0);

    for (int s = 0; s < nsteps; s++) {
        const int buf = s & 1;
        wait_async0();        // my DMA into sA/sB[buf] is done
        __syncthreads();      // => everyone's DMA is done; prev reads retired
        if (s + 1 < nsteps) stage(buf ^ 1, (s + 1) * BK);

        const unsigned short* cA = sA[buf];
        const unsigned short* cB = sB[buf];

        // A fragments: lane holds row M=l16; K 0..15 (half 0) / 16..31 (half 1)
        F a[2], b[4];
        for (int i = 0; i < 2; i++) {
            int row = wm * 32 + i * 16 + l16;
            int g0  = half * 16;
            a[i].u[0] = *(const uint4*)(&cA[row * LSTR + g0]);
            a[i].u[1] = *(const uint4*)(&cA[row * LSTR + g0 + 8]);
        }
        // B fragments: lane holds column N=l16; K 0..15 (half 0) / 16..31 (half 1)
        for (int j = 0; j < 4; j++) {
            int col = wn * 64 + j * 16 + l16;
            int g0  = half * 16;
            b[j].u[0] = *(const uint4*)(&cB[col * LSTR + g0]);
            b[j].u[1] = *(const uint4*)(&cB[col * LSTR + g0 + 8]);
        }
        for (int i = 0; i < 2; i++)
            for (int j = 0; j < 4; j++)
                acc[i][j] = __builtin_amdgcn_wmma_f32_16x16x32_bf16(
                    false, a[i].v, false, b[j].v, (short)0, acc[i][j], false, false);
    }

    // epilogue: C layout — VGPR r holds row (half*8 + r), lane = column l16
    for (int i = 0; i < 2; i++) {
        int rbase = m0 + wm * 32 + i * 16 + half * 8;
        for (int j = 0; j < 4; j++) {
            int n = n0 + wn * 64 + j * 16 + l16;
            if (n < N) {
                for (int r = 0; r < 8; r++) {
                    float v = acc[i][j][r];
                    if constexpr (OUT_F32)
                        ((float*)C)[(size_t)(rbase + r) * ldc + n] = v;
                    else
                        ((unsigned short*)C)[(size_t)(rbase + r) * ldc + n] = f32_to_bf16(v);
                }
            }
        }
    }
}

// ---------------------------------------------------------------------------
// Fused flash attention. One wave = 16 query rows of one (b,h).
// q:      (B*S, H*128) bf16
// comp:   (B*S, 576)  bf16  [0:512 kv_c | 512:576 k_rope]
// kvexp:  (B*S, H*192) bf16 [per head: 0:128 v | 128:192 k_nope]
// mask:   (S,S) fp32
// outb:   (B*S, H*128) bf16
// k[t,d] = d<64 ? k_rope[t,d] : k_nope[t,d-64]
// ---------------------------------------------------------------------------
__global__ __launch_bounds__(128)
void mla_flash_attn(const unsigned short* __restrict__ qb,
                    const unsigned short* __restrict__ comp,
                    const unsigned short* __restrict__ kvexp,
                    const float* __restrict__ mask,
                    unsigned short* __restrict__ outb)
{
    constexpr float scale = 0.08838834764831845f;   // 128^-0.5
    __shared__ __align__(16) unsigned short pTile[4][16 * 32];

    const int tid  = threadIdx.x;
    const int lane = tid & 31;
    const int wv   = tid >> 5;
    const int half = lane >> 4;
    const int l16  = lane & 15;

    const int unit = blockIdx.x * 4 + wv;
    const int QB   = CS / 16;                       // 128 q-blocks per (b,h)
    const int qblk = unit % QB;
    const int h    = (unit / QB) % CH;
    const int b    = unit / (QB * CH);

    const int    s0      = qblk * 16;
    const size_t rowBase = (size_t)b * CS;

    union F  { uint4 u[2];            v16bf v; };
    union FS { unsigned short s[16];  v16bf v; };

    // Q fragments (d = 0..127), resident for the whole kernel
    F qf[4];
    {
        const unsigned short* qrow = qb + (rowBase + s0 + l16) * (CH * CQD) + h * CQD;
        for (int kk = 0; kk < 4; kk++) {
            int g = half * 16;
            qf[kk].u[0] = *(const uint4*)(qrow + kk * 32 + g);
            qf[kk].u[1] = *(const uint4*)(qrow + kk * 32 + g + 8);
        }
    }

    v8f o[8];
    for (int n = 0; n < 8; n++)
        for (int e = 0; e < 8; e++) o[n][e] = 0.0f;
    float mrow[8], lrow[8];
    for (int r = 0; r < 8; r++) { mrow[r] = -1e30f; lrow[r] = 0.0f; }

    for (int t0 = 0; t0 < CS; t0 += 32) {
        // ---- scores: two 16x16 tiles, S = Q(16x128) . K^T(128x32) ----
        v8f sc[2];
        for (int tt = 0; tt < 2; tt++) for (int e = 0; e < 8; e++) sc[tt][e] = 0.0f;
        for (int tt = 0; tt < 2; tt++) {
            int t = t0 + tt * 16 + l16;             // this lane's key column
            const unsigned short* ropeRow = comp  + (rowBase + t) * 576 + 512;
            const unsigned short* nopeRow = kvexp + (rowBase + t) * (CH * 192) + h * 192 + 128;
            for (int kk = 0; kk < 4; kk++) {
                F bf;
                const unsigned short* src = (kk < 2) ? (ropeRow + kk * 32)
                                                     : (nopeRow + (kk - 2) * 32);
                int g = half * 16;
                bf.u[0] = *(const uint4*)(src + g);
                bf.u[1] = *(const uint4*)(src + g + 8);
                sc[tt] = __builtin_amdgcn_wmma_f32_16x16x32_bf16(
                    false, qf[kk].v, false, bf.v, (short)0, sc[tt], false, false);
            }
        }
        // ---- scale + mask ----
        float p[2][8];
        for (int tt = 0; tt < 2; tt++) {
            int t = t0 + tt * 16 + l16;
            for (int r = 0; r < 8; r++) {
                int srow = s0 + half * 8 + r;
                p[tt][r] = sc[tt][r] * scale + mask[(size_t)srow * CS + t];
            }
        }
        // ---- online softmax: row reductions across 16-lane halves ----
        float mnew[8], alpha[8];
        for (int r = 0; r < 8; r++) {
            float v = fmaxf(p[0][r], p[1][r]);
            for (int m = 1; m < 16; m <<= 1) v = fmaxf(v, __shfl_xor(v, m, 32));
            mnew[r]  = fmaxf(mrow[r], v);
            alpha[r] = __expf(mrow[r] - mnew[r]);
        }
        for (int r = 0; r < 8; r++) {
            p[0][r] = __expf(p[0][r] - mnew[r]);
            p[1][r] = __expf(p[1][r] - mnew[r]);
            float s = p[0][r] + p[1][r];
            for (int m = 1; m < 16; m <<= 1) s += __shfl_xor(s, m, 32);
            lrow[r] = lrow[r] * alpha[r] + s;
            mrow[r] = mnew[r];
        }
        for (int n = 0; n < 8; n++)
            for (int r = 0; r < 8; r++) o[n][r] *= alpha[r];

        // ---- restage P (C layout -> A layout) through per-wave LDS ----
        unsigned short* pt = pTile[wv];
        for (int tt = 0; tt < 2; tt++)
            for (int r = 0; r < 8; r++) {
                int row = half * 8 + r;
                pt[row * 32 + tt * 16 + l16] = f32_to_bf16(p[tt][r]);
            }
        F pf;                                        // A frag: lane = row l16
        {
            int g = half * 16;
            pf.u[0] = *(const uint4*)(&pt[l16 * 32 + g]);
            pf.u[1] = *(const uint4*)(&pt[l16 * 32 + g + 8]);
        }

        // ---- O += P(16x32) . V(32x128), V gathered per column (L2 resident) ----
        for (int n = 0; n < 8; n++) {
            FS vf;
            int vd = n * 16 + l16;
            const unsigned short* vcol =
                kvexp + (rowBase + t0 + half * 16) * (CH * 192) + h * 192 + vd;
            for (int i = 0; i < 16; i++)
                vf.s[i] = vcol[(size_t)i * (CH * 192)];
            o[n] = __builtin_amdgcn_wmma_f32_16x16x32_bf16(
                false, pf.v, false, vf.v, (short)0, o[n], false, false);
        }
    }

    // ---- normalize and emit bf16 ----
    for (int n = 0; n < 8; n++) {
        int col = h * CVD + n * 16 + l16;
        for (int r = 0; r < 8; r++) {
            int srow = s0 + half * 8 + r;
            outb[(rowBase + srow) * (CH * CVD) + col] = f32_to_bf16(o[n][r] / lrow[r]);
        }
    }
}

// ---------------------------------------------------------------------------
// Host orchestration
// ---------------------------------------------------------------------------
extern "C" void kernel_launch(void* const* d_in, const int* in_sizes, int n_in,
                              void* d_out, int out_size, void* d_ws, size_t ws_size,
                              hipStream_t stream) {
    const float* x    = (const float*)d_in[0];   // (B,S,E)
    const float* mask = (const float*)d_in[1];   // (1,1,S,S)
    const float* Wkvd = (const float*)d_in[2];   // (576, E)
    const float* Wkvu = (const float*)d_in[3];   // (3072, 512)
    const float* Wq   = (const float*)d_in[4];   // (2048, E)
    const float* Wout = (const float*)d_in[5];   // (E, 2048)
    float* out = (float*)d_out;

    const int M = CB * CS;                       // 4096 rows

    char* wp = (char*)d_ws;
    auto alloc = [&](size_t bytes) -> unsigned short* {
        unsigned short* p = (unsigned short*)wp;
        wp += (bytes + 255) & ~(size_t)255;
        return p;
    };
    unsigned short* xb    = alloc((size_t)M * CE * 2);
    unsigned short* Wqb   = alloc((size_t)2048 * CE * 2);
    unsigned short* Wkvdb = alloc((size_t)576 * CE * 2);
    unsigned short* Wkvub = alloc((size_t)3072 * 512 * 2);
    unsigned short* Woutb = alloc((size_t)CE * 2048 * 2);
    unsigned short* qbuf  = alloc((size_t)M * 2048 * 2);
    unsigned short* compb = alloc((size_t)M * 576 * 2);
    unsigned short* kvexb = alloc((size_t)M * 3072 * 2);
    unsigned short* attnb = alloc((size_t)M * 2048 * 2);

    auto cvt = [&](const float* s, unsigned short* d, long n) {
        long n4 = n / 4;
        cvt_f32_bf16_x4<<<dim3((unsigned)((n4 + 255) / 256)), dim3(256), 0, stream>>>(s, d, n4);
    };
    cvt(x,    xb,    (long)M * CE);
    cvt(Wq,   Wqb,   (long)2048 * CE);
    cvt(Wkvd, Wkvdb, (long)576 * CE);
    cvt(Wkvu, Wkvub, (long)3072 * 512);
    cvt(Wout, Woutb, (long)CE * 2048);

    dim3 blk(256);
    // q = x @ W_q^T                  (4096 x 2048, K=2048)
    gemm_bf16_nt<false><<<dim3(2048 / 128, M / 128), blk, 0, stream>>>(
        xb, Wqb, qbuf, M, 2048, CE, CE, CE, 2048);
    // compressed = x @ W_kv_down^T   (4096 x 576, K=2048)
    gemm_bf16_nt<false><<<dim3((576 + 127) / 128, M / 128), blk, 0, stream>>>(
        xb, Wkvdb, compb, M, 576, CE, CE, CE, 576);
    // kv_exp = kv_c @ W_kv_up^T      (4096 x 3072, K=512, A strided in compressed)
    gemm_bf16_nt<false><<<dim3(3072 / 128, M / 128), blk, 0, stream>>>(
        compb, Wkvub, kvexb, M, 3072, 512, 576, 512, 3072);
    // fused flash attention          (B*H*S/16 = 4096 wave-units, 4 waves/block)
    mla_flash_attn<<<dim3((CB * CH * (CS / 16)) / 4), dim3(128), 0, stream>>>(
        qbuf, compb, kvexb, mask, attnb);
    // final = attn_out @ W_out^T     (4096 x 2048, K=2048, fp32 output)
    gemm_bf16_nt<true><<<dim3(2048 / 128, M / 128), blk, 0, stream>>>(
        attnb, Woutb, out, M, 2048, 2048, 2048, 2048, 2048);
}
